// ADiGCNConv_15350213116045
// MI455X (gfx1250) — matrix operand order repaired and checked by reading, not compiled, not run on hardware
//
#include <hip/hip_runtime.h>
#include <hip/hip_bf16.h>
#include <cstdint>

// Problem constants (match reference)
#define NN 50000
#define EE 800000
#define DD 128
#define ALPHA_C 0.5f

typedef __attribute__((ext_vector_type(16))) __bf16        v16bf;
typedef __attribute__((ext_vector_type(8)))  float         v8f;
typedef __attribute__((ext_vector_type(4)))  float         v4f;
typedef __attribute__((ext_vector_type(4)))  unsigned int  v4u;
typedef __attribute__((ext_vector_type(8)))  int           v8i;
typedef __attribute__((ext_vector_type(4)))  int           v4i;

// LDS (addrspace 3) pointer for volatile ds_load access.
typedef const volatile __attribute__((address_space(3))) v4f* lds_v4f_ptr;

// ---------------------------------------------------------------------------
// 1. zero workspace accumulators
// ---------------------------------------------------------------------------
__global__ void zero_f32_kernel(float* __restrict__ p, long n) {
    long i = (long)blockIdx.x * blockDim.x + threadIdx.x;
    long stride = (long)gridDim.x * blockDim.x;
    float4* p4 = (float4*)p;
    long n4 = n >> 2;
    float4 z = make_float4(0.f, 0.f, 0.f, 0.f);
    for (long j = i; j < n4; j += stride) p4[j] = z;
}

// ---------------------------------------------------------------------------
// 2. degree counting from edge list (float counts, like segment_sum of ones)
// ---------------------------------------------------------------------------
__global__ void degree_kernel(const int* __restrict__ row, const int* __restrict__ col,
                              float* __restrict__ out_deg, float* __restrict__ in_deg, int e) {
    int i = blockIdx.x * blockDim.x + threadIdx.x;
    if (i < e) {
        atomicAdd(&out_deg[row[i]], 1.0f);
        atomicAdd(&in_deg[col[i]], 1.0f);
    }
}

// ---------------------------------------------------------------------------
// 3. D^{-1/2} normalizers
// ---------------------------------------------------------------------------
__global__ void invsqrt_kernel(const float* __restrict__ od, const float* __restrict__ id,
                               float* __restrict__ oi, float* __restrict__ ii, int n) {
    int i = blockIdx.x * blockDim.x + threadIdx.x;
    if (i < n) {
        float a = od[i], b = id[i];
        oi[i] = (a > 0.f) ? rsqrtf(a) : 0.f;
        ii[i] = (b > 0.f) ? rsqrtf(b) : 0.f;
    }
}

// ---------------------------------------------------------------------------
// 4. SpMM scatter: one wave32 per edge, lane handles 4 contiguous features.
//    out_nei[row] += w * x[col];  in_nei[col] += w * x[row]
// ---------------------------------------------------------------------------
__global__ __launch_bounds__(256) void spmm_kernel(
    const int* __restrict__ row, const int* __restrict__ col,
    const float* __restrict__ x,
    const float* __restrict__ oi, const float* __restrict__ ii,
    float* __restrict__ out_nei, float* __restrict__ in_nei, int e) {
    int wave = (int)((blockIdx.x * (long)blockDim.x + threadIdx.x) >> 5);
    int lane = threadIdx.x & 31;
    if (wave >= e) return;
    int r = row[wave], c = col[wave];
    float w = oi[r] * ii[c];
    float4 xc = ((const float4*)(x + (long)c * DD))[lane];
    float4 xr = ((const float4*)(x + (long)r * DD))[lane];
    float* on = out_nei + (long)r * DD + lane * 4;
    float* ip = in_nei + (long)c * DD + lane * 4;
    atomicAdd(on + 0, w * xc.x); atomicAdd(on + 1, w * xc.y);
    atomicAdd(on + 2, w * xc.z); atomicAdd(on + 3, w * xc.w);
    atomicAdd(ip + 0, w * xr.x); atomicAdd(ip + 1, w * xr.y);
    atomicAdd(ip + 2, w * xr.z); atomicAdd(ip + 3, w * xr.w);
}

// ---------------------------------------------------------------------------
// 5. gate kernel: one wave32 per node; two 128-wide dot products, 2-way
//    softmax, degree-mask gating. Writes C_in/C_out to d_out tail + workspace.
// ---------------------------------------------------------------------------
__global__ __launch_bounds__(256) void gate_kernel(
    const float* __restrict__ x,
    const float* __restrict__ out_nei, const float* __restrict__ in_nei,
    const int* __restrict__ out_degree, const int* __restrict__ in_degree,
    const float* __restrict__ out_table, const float* __restrict__ in_table,
    const float* __restrict__ W_out_f, const float* __restrict__ W_in_f,
    const float* __restrict__ b_out_f, const float* __restrict__ b_in_f,
    const float* __restrict__ out_mask, const float* __restrict__ out_mask_b,
    const float* __restrict__ in_mask, const float* __restrict__ in_mask_b,
    float* __restrict__ Co, float* __restrict__ Ci,
    float* __restrict__ tail_ci, float* __restrict__ tail_co, int n) {
    int node = (int)((blockIdx.x * (long)blockDim.x + threadIdx.x) >> 5);
    int lane = threadIdx.x & 31;
    if (node >= n) return;
    long base = (long)node * DD;
    float4 xx = ((const float4*)(x + base))[lane];
    float4 on = ((const float4*)(out_nei + base))[lane];
    float4 in = ((const float4*)(in_nei + base))[lane];
    float4 to = ((const float4*)(out_table + (long)out_degree[node] * DD))[lane];
    float4 ti = ((const float4*)(in_table + (long)in_degree[node] * DD))[lane];
    float4 wo = ((const float4*)W_out_f)[lane];
    float4 wi = ((const float4*)W_in_f)[lane];
    float so = (on.x - xx.x + to.x) * wo.x + (on.y - xx.y + to.y) * wo.y +
               (on.z - xx.z + to.z) * wo.z + (on.w - xx.w + to.w) * wo.w;
    float si = (in.x - xx.x + ti.x) * wi.x + (in.y - xx.y + ti.y) * wi.y +
               (in.z - xx.z + ti.z) * wi.z + (in.w - xx.w + ti.w) * wi.w;
#pragma unroll
    for (int off = 16; off > 0; off >>= 1) {
        so += __shfl_xor(so, off, 32);
        si += __shfl_xor(si, off, 32);
    }
    if (lane == 0) {
        float co = so + b_out_f[0];
        float ci = si + b_in_f[0];
        float m = fmaxf(co, ci);
        float eo = __expf(co - m), ei = __expf(ci - m);
        float inv = 1.0f / (eo + ei);
        float C0 = eo * inv, C1 = ei * inv;
        float gco = C0 * out_mask[node] + out_mask_b[node];
        float gci = C1 * in_mask[node] + in_mask_b[node];
        Co[node] = gco;
        Ci[node] = gci;
        tail_co[node] = gco;
        tail_ci[node] = gci;
    }
}

// ---------------------------------------------------------------------------
// 6. fused triple-GEMM + gating epilogue with bf16 WMMA.
//    A tiles (16 rows x 128 f32, shared by all 8 waves of the block) are
//    staged into LDS once per block with the Tensor Data Mover
//    (tensor_load_to_lds, TENSORcnt), removing the 8x redundant L2 reads.
//    LDS reads use explicit addrspace(3) volatile pointers so they lower to
//    ds_load_b128 (generic volatile pointers lowered to flat SYS-scope loads;
//    non-volatile loads were folded away since the TDM writes LDS outside the
//    compiler's alias view).
//    Fragment layouts (wave32):
//      A (16x32 bf16): lane L -> row m0+(L&15), group g=L>>4
//        halves 0..7  <- K = kk + 16g + [0..7]      (contiguous)
//        halves 8..15 <- K = kk + 16g + 8 + [0..7]  (contiguous)
//      B (32x16 bf16): lane L -> col n0+(L&15), group g=L>>4
//        halves 0..15 <- K = kk + 16g + [0..15]     (contiguous; B[k,n]=W[n,k])
// ---------------------------------------------------------------------------

// Build a TDM D# for a 2D 16x128 f32 tile (row-major, row stride 128) and
// issue TENSOR_LOAD_TO_LDS.  Descriptor packing per CDNA5 ISA ch.8:
//   g0: [1:0]=count=1, [63:32]=lds_addr, [120:64]=global_addr, [127:126]=type=2
//   g1: [17:16]=data_size(2 -> 4B), [79:48]=tensor_dim0=128,
//       [111:80]=tensor_dim1=16, [127:112]=tile_dim0=128,
//       [143:128]=tile_dim1=16, [207:160]=tensor_dim0_stride=128
//   g2/g3: zero (2D tensor); trailing group also zero.
__device__ inline void tdm_load_tile16x128(const float* gptr, unsigned lds_byte_off) {
    unsigned long long ga = (unsigned long long)(uintptr_t)gptr;
    v4u g0;
    g0[0] = 1u;                                   // count=1 (valid user D#)
    g0[1] = lds_byte_off;                         // lds_addr (bytes)
    g0[2] = (unsigned)(ga & 0xFFFFFFFFu);         // global_addr[31:0]
    g0[3] = (unsigned)((ga >> 32) & 0x01FFFFFFu)  // global_addr[56:32]
            | 0x80000000u;                        // type=2 ("image")
    v8i g1;
    g1[0] = 0x00020000;       // workgroup_mask=0, data_size=2 (4 bytes)
    g1[1] = (DD << 16);       // tensor_dim0[15:0] = 128
    g1[2] = (16 << 16);       // tensor_dim0[31:16]=0, tensor_dim1[15:0]=16
    g1[3] = (DD << 16);       // tensor_dim1[31:16]=0, tile_dim0=128
    g1[4] = 16;               // tile_dim1=16, tile_dim2=0
    g1[5] = DD;               // tensor_dim0_stride[31:0] = 128
    g1[6] = 0;                // stride hi / tensor_dim1_stride lo
    g1[7] = 0;
    v4i g2 = {0, 0, 0, 0};
    v4i g3 = {0, 0, 0, 0};
    v8i g4 = {0, 0, 0, 0, 0, 0, 0, 0};
    // 6-arg form on this toolchain: (g0, g1, g2, g3, g4, cpol)
    __builtin_amdgcn_tensor_load_to_lds(g0, g1, g2, g3, g4, 0);
}

// A fragment from the TDM-written LDS tile. byte_off is the LDS byte address
// of this lane's row at the current k-step (allocation starts at offset 0).
// Volatile addrspace(3) vector loads -> ds_load_b128.
__device__ inline v16bf load_a_frag_lds(unsigned byte_off, int g) {
    lds_v4f_ptr p0 = (lds_v4f_ptr)(byte_off + 64u * (unsigned)g);
    lds_v4f_ptr p1 = (lds_v4f_ptr)(byte_off + 32u + 64u * (unsigned)g);
    v4f a0 = p0[0];
    v4f a1 = p0[1];
    v4f b0 = p1[0];
    v4f b1 = p1[1];
    float t[16] = {a0[0], a0[1], a0[2], a0[3], a1[0], a1[1], a1[2], a1[3],
                   b0[0], b0[1], b0[2], b0[3], b1[0], b1[1], b1[2], b1[3]};
    v16bf r;
#pragma unroll
    for (int i = 0; i < 16; ++i) r[i] = (__bf16)t[i];
    return r;
}

__device__ inline v16bf load_b_frag(const float* __restrict__ p) {
    float4 a0 = *(const float4*)(p);
    float4 a1 = *(const float4*)(p + 4);
    float4 a2 = *(const float4*)(p + 8);
    float4 a3 = *(const float4*)(p + 12);
    float t[16] = {a0.x, a0.y, a0.z, a0.w, a1.x, a1.y, a1.z, a1.w,
                   a2.x, a2.y, a2.z, a2.w, a3.x, a3.y, a3.z, a3.w};
    v16bf r;
#pragma unroll
    for (int i = 0; i < 16; ++i) r[i] = (__bf16)t[i];
    return r;
}

__global__ __launch_bounds__(256) void wmma_out_kernel(
    const float* __restrict__ x,
    const float* __restrict__ out_nei, const float* __restrict__ in_nei,
    const float* __restrict__ W_src, const float* __restrict__ b_src,
    const float* __restrict__ W_dst, const float* __restrict__ b_dst,
    const float* __restrict__ W_fc, const float* __restrict__ b_fc,
    const float* __restrict__ Co, const float* __restrict__ Ci,
    float* __restrict__ out, int n) {
    // Single static LDS allocation -> tile offsets within LDS start at 0.
    // Accessed via the TDM (write) and addrspace(3) offsets (read); the asm
    // use below pins the allocation so group_segment size stays 24 KB.
    __shared__ float smem[3 * 16 * DD];           // 24 KB: [x | out_nei | in_nei]
    asm volatile("" : : "v"((const void*)smem) : "memory");

    int m0 = blockIdx.x * 16;
    if (m0 + 16 > n) return;                      // N % 16 == 0 for this problem
    int wv = threadIdx.x >> 5;                    // 8 waves -> 8 N-tiles of 16 cols
    int lane = threadIdx.x & 31;
    int n0 = wv * 16;
    int mL = lane & 15;
    int g = lane >> 4;

    // Wave 0 DMAs the three shared A tiles into LDS via the TDM, waits for
    // TENSORcnt, then the block barrier publishes LDS to all waves.
    if (wv == 0) {
        tdm_load_tile16x128(x + (long)m0 * DD, 0u);
        tdm_load_tile16x128(out_nei + (long)m0 * DD, 16u * DD * 4u);
        tdm_load_tile16x128(in_nei + (long)m0 * DD, 2u * 16u * DD * 4u);
        __builtin_amdgcn_s_wait_tensorcnt(0);
    }
    __syncthreads();

    // LDS byte offsets of this lane's A rows (tiles: x @0, out_nei @8KB,
    // in_nei @16KB; row stride = 128 floats = 512 B).
    unsigned aoff_x = (unsigned)(mL * DD) * 4u;
    unsigned aoff_o = (unsigned)((16 + mL) * DD) * 4u;
    unsigned aoff_i = (unsigned)((32 + mL) * DD) * 4u;
    const float* brow_s = W_src + (long)(n0 + mL) * DD + 16 * g;
    const float* brow_d = W_dst + (long)(n0 + mL) * DD + 16 * g;
    const float* brow_f = W_fc + (long)(n0 + mL) * DD + 16 * g;

    v8f acc_s = {}; v8f acc_d = {}; v8f acc_f = {};
#pragma unroll
    for (int kk = 0; kk < DD; kk += 32) {
        v16bf a_o = load_a_frag_lds(aoff_o + kk * 4u, g);
        v16bf a_i = load_a_frag_lds(aoff_i + kk * 4u, g);
        v16bf a_x = load_a_frag_lds(aoff_x + kk * 4u, g);
        v16bf b_s = load_b_frag(brow_s + kk);
        v16bf b_d = load_b_frag(brow_d + kk);
        v16bf b_f = load_b_frag(brow_f + kk);
        acc_s = __builtin_amdgcn_wmma_f32_16x16x32_bf16(
            false, a_o, false, b_s, (short)0, acc_s, false, false);
        acc_d = __builtin_amdgcn_wmma_f32_16x16x32_bf16(
            false, a_i, false, b_d, (short)0, acc_d, false, false);
        acc_f = __builtin_amdgcn_wmma_f32_16x16x32_bf16(
            false, a_x, false, b_f, (short)0, acc_f, false, false);
    }

    // Epilogue: D layout (f32 16x16): lane L holds col n0+(L&15);
    // VGPR v holds row m0 + v + 8*(L>>4).
    int ncol = n0 + mL;
    float bs = b_src[ncol], bd = b_dst[ncol], bf = b_fc[ncol];
#pragma unroll
    for (int v = 0; v < 8; ++v) {
        int m = m0 + v + 8 * g;
        float co = Co[m], ci = Ci[m];
        out[(long)m * DD + ncol] =
            co * (acc_s[v] + bs) + ci * (acc_d[v] + bd) + ALPHA_C * (acc_f[v] + bf);
    }
}

// ---------------------------------------------------------------------------
// launcher
// ---------------------------------------------------------------------------
extern "C" void kernel_launch(void* const* d_in, const int* in_sizes, int n_in,
                              void* d_out, int out_size, void* d_ws, size_t ws_size,
                              hipStream_t stream) {
    const float* x            = (const float*)d_in[0];
    const int*   edge_index   = (const int*)d_in[1];
    const int*   in_degree    = (const int*)d_in[2];
    const int*   out_degree   = (const int*)d_in[3];
    const float* out_mask     = (const float*)d_in[4];
    const float* out_mask_b   = (const float*)d_in[5];
    const float* in_mask      = (const float*)d_in[6];
    const float* in_mask_b    = (const float*)d_in[7];
    const float* W_src        = (const float*)d_in[8];
    const float* b_src        = (const float*)d_in[9];
    const float* W_dst        = (const float*)d_in[10];
    const float* b_dst        = (const float*)d_in[11];
    const float* W_out_f      = (const float*)d_in[12];
    const float* b_out_f      = (const float*)d_in[13];
    const float* W_in_f       = (const float*)d_in[14];
    const float* b_in_f       = (const float*)d_in[15];
    const float* W_fc         = (const float*)d_in[16];
    const float* b_fc         = (const float*)d_in[17];
    const float* in_table     = (const float*)d_in[18];
    const float* out_table    = (const float*)d_in[19];

    const int N = in_sizes[0] / DD;       // 50000
    const int E = in_sizes[1] / 2;        // 800000
    const int* row = edge_index;
    const int* col = edge_index + E;

    // workspace layout (floats)
    float* ws       = (float*)d_ws;
    float* out_deg  = ws;                       // N
    float* in_deg   = out_deg + N;              // N
    float* out_nei  = in_deg + N;               // N*D
    float* in_nei   = out_nei + (long)N * DD;   // N*D
    float* o_inv    = in_nei + (long)N * DD;    // N
    float* i_inv    = o_inv + N;                // N
    float* Co       = i_inv + N;                // N
    float* Ci       = Co + N;                   // N

    float* out      = (float*)d_out;            // N*D
    float* tail_ci  = out + (long)N * DD;       // N (C_in)
    float* tail_co  = tail_ci + N;              // N (C_out)

    // 1. zero accumulators (out_deg, in_deg, out_nei, in_nei are contiguous)
    zero_f32_kernel<<<4096, 256, 0, stream>>>(out_deg, (long)N * (2 + 2 * DD));
    // 2. degrees
    degree_kernel<<<(E + 255) / 256, 256, 0, stream>>>(row, col, out_deg, in_deg, E);
    // 3. normalizers
    invsqrt_kernel<<<(N + 255) / 256, 256, 0, stream>>>(out_deg, in_deg, o_inv, i_inv, N);
    // 4. SpMM scatter (wave per edge)
    spmm_kernel<<<(E + 7) / 8, 256, 0, stream>>>(row, col, x, o_inv, i_inv,
                                                 out_nei, in_nei, E);
    // 5. gates (wave per node)
    gate_kernel<<<(N + 7) / 8, 256, 0, stream>>>(
        x, out_nei, in_nei, out_degree, in_degree, out_table, in_table,
        W_out_f, W_in_f, b_out_f, b_in_f,
        out_mask, out_mask_b, in_mask, in_mask_b,
        Co, Ci, tail_ci, tail_co, N);
    // 6. fused triple GEMM + gated combine (TDM-staged A tiles + bf16 WMMA)
    wmma_out_kernel<<<(N + 15) / 16, 256, 0, stream>>>(
        x, out_nei, in_nei, W_src, b_src, W_dst, b_dst, W_fc, b_fc,
        Co, Ci, out, N);
}
